// GNNTransformer_68496138437287
// MI455X (gfx1250) — compile-verified, matrix operand order checked by reading, not run on hardware
//
#include <hip/hip_runtime.h>
#include <hip/hip_bf16.h>
#include <math.h>

typedef __attribute__((ext_vector_type(16))) _Float16 v16h;
typedef __attribute__((ext_vector_type(8)))  _Float16 v8h;
typedef __attribute__((ext_vector_type(8)))  float    v8f;

#define Bn 8
#define Tn 32
#define Nn 64
#define Fd 6
#define Dd 128
#define Hh 4
#define Cc 32
#define En 4032            // N*(N-1)
#define NT (Bn*Tn*Nn)      // 16384 tokens
#define SEQ (Bn*Nn)        // 512 temporal sequences
#define SCALE 0.17677669529663687f   // 1/sqrt(32)

__device__ __forceinline__ float gelu_exact(float x) {
  return 0.5f * x * (1.0f + erff(x * 0.70710678118654752f));
}

// LDS byte offset of a __shared__ object: generic LDS-aperture pointers carry
// the LDS offset in the low 32 bits (ISA 10.2).
__device__ __forceinline__ unsigned lds_off32(const void* p) {
  return (unsigned)(uintptr_t)p;
}

// Build a f16 WMMA A/B fragment from a contiguous LDS row: the per-lane k-map
// is two contiguous 8-half runs at offsets lk0 and lk0+16 (lk0 = lane>=16 ? 8:0),
// i.e. exactly two 16-byte ds_load_b128s + a register shuffle.
__device__ __forceinline__ v16h frag16(const _Float16* __restrict__ row, int lk0) {
  v8h lo = *(const v8h*)(row + lk0);
  v8h hi = *(const v8h*)(row + lk0 + 16);
  return __builtin_shufflevector(lo, hi, 0, 1, 2, 3, 4, 5, 6, 7,
                                 8, 9, 10, 11, 12, 13, 14, 15);
}

// ---------------------------------------------------------------------------
// WMMA GEMM: out[M,N] = act(A[M,K] @ W[K,N] + bias), fp32 I/O, f16 mul/f32 acc.
// 256 threads = 8 waves; block tile 64x64; wave tile 16x32 (one A fragment,
// two B fragments / accumulators). LDS double-buffered f16 staging; B staged
// transposed so fragment reads are contiguous. M%64==0, K%32==0 required.
// NGUARD=false requires N%64==0 (all hot GEMMs); NGUARD=true handles any N.
// ---------------------------------------------------------------------------
template <bool NGUARD>
__global__ __launch_bounds__(256) void k_gemm(
    const float* __restrict__ A, int lda,
    const float* __restrict__ Wt, const float* __restrict__ bias,
    float* __restrict__ out, int ldo, int M, int N, int K, int act) {
  const int tid = threadIdx.x;
  const int lane = tid & 31, wave = tid >> 5;
  const int nBlocksN = (N + 63) >> 6;
  const int bm = (blockIdx.x / nBlocksN) << 6;
  const int bn = (blockIdx.x % nBlocksN) << 6;

  __shared__ _Float16 lA[2][64][32];     // [buf][m][k]
  __shared__ _Float16 lB[2][64][32];     // [buf][n][k]  (transposed)

  const int mband = (wave & 3) << 4;
  const int nhalf = (wave >> 2) << 5;
  const int mloc = mband + (lane & 15);
  const int lk0 = (lane & 16) ? 8 : 0;

  // staging indices (per thread, fixed across K-steps)
  const int sa_m = tid >> 2;             // 0..63
  const int sa_k = (tid & 3) << 3;       // 0,8,16,24
  const int sb_k = tid >> 3;             // 0..31
  const int sb_n = (tid & 7) << 3;       // 0..56

  v8f acc0 = {}, acc1 = {};

  auto stage = [&](int buf, int kb) {
    // A tile: 64x32 f32 -> f16, vectorized float4 loads, one b128 LDS store.
    {
      const float* src = A + (size_t)(bm + sa_m) * lda + kb + sa_k;
      const float4 f0 = *(const float4*)(src);
      const float4 f1 = *(const float4*)(src + 4);
      v8h h;
      h[0] = (_Float16)f0.x; h[1] = (_Float16)f0.y;
      h[2] = (_Float16)f0.z; h[3] = (_Float16)f0.w;
      h[4] = (_Float16)f1.x; h[5] = (_Float16)f1.y;
      h[6] = (_Float16)f1.z; h[7] = (_Float16)f1.w;
      *(v8h*)&lA[buf][sa_m][sa_k] = h;
    }
    // B tile: 32x64 f32 -> f16 transposed into LDS.
    {
      const float* src = Wt + (size_t)(kb + sb_k) * N + bn + sb_n;
      if (!NGUARD) {
        const float4 f0 = *(const float4*)(src);
        const float4 f1 = *(const float4*)(src + 4);
        lB[buf][sb_n + 0][sb_k] = (_Float16)f0.x;
        lB[buf][sb_n + 1][sb_k] = (_Float16)f0.y;
        lB[buf][sb_n + 2][sb_k] = (_Float16)f0.z;
        lB[buf][sb_n + 3][sb_k] = (_Float16)f0.w;
        lB[buf][sb_n + 4][sb_k] = (_Float16)f1.x;
        lB[buf][sb_n + 5][sb_k] = (_Float16)f1.y;
        lB[buf][sb_n + 6][sb_k] = (_Float16)f1.z;
        lB[buf][sb_n + 7][sb_k] = (_Float16)f1.w;
      } else {
#pragma unroll
        for (int j = 0; j < 8; ++j) {
          const int n = bn + sb_n + j;
          lB[buf][sb_n + j][sb_k] = (n < N) ? (_Float16)src[j] : (_Float16)0.0f;
        }
      }
    }
  };

  const int nsteps = K >> 5;
  stage(0, 0);
  for (int s = 0; s < nsteps; ++s) {
    __syncthreads();
    if (s + 1 < nsteps) stage((s + 1) & 1, (s + 1) << 5);
    const int buf = s & 1;
    const v16h a = frag16(&lA[buf][mloc][0], lk0);
    const v16h b0 = frag16(&lB[buf][nhalf + (lane & 15)][0], lk0);
    const v16h b1 = frag16(&lB[buf][nhalf + 16 + (lane & 15)][0], lk0);
    acc0 = __builtin_amdgcn_wmma_f32_16x16x32_f16(false, a, false, b0,
                                                  (short)0, acc0, false, false);
    acc1 = __builtin_amdgcn_wmma_f32_16x16x32_f16(false, a, false, b1,
                                                  (short)0, acc1, false, false);
  }

  const v8f accs[2] = {acc0, acc1};
#pragma unroll
  for (int nt = 0; nt < 2; ++nt) {
    const int ncol = bn + nhalf + nt * 16 + (lane & 15);
    const bool nok = !NGUARD || (ncol < N);
    const float bv = (bias && nok) ? bias[ncol] : 0.0f;
#pragma unroll
    for (int r = 0; r < 8; ++r) {
      const int i = bm + mband + r + ((lane & 16) ? 8 : 0);
      float x = accs[nt][r] + bv;
      if (act == 1) x = gelu_exact(x);
      if (nok) out[(size_t)i * ldo + ncol] = x;
    }
  }
}

// ---------------------------------------------------------------------------
// Input projection: h = frames @ proj_W + proj_b  (K=6, scalar FMA path)
// ---------------------------------------------------------------------------
__global__ void k_proj(const float* __restrict__ in, const float* __restrict__ W,
                       const float* __restrict__ bias, float* __restrict__ h) {
  const size_t idx = (size_t)blockIdx.x * blockDim.x + threadIdx.x;
  if (idx >= (size_t)NT * Dd) return;
  const int d = (int)(idx & (Dd - 1));
  const size_t node = idx >> 7;
  float s = bias[d];
#pragma unroll
  for (int f = 0; f < Fd; ++f) s += in[node * Fd + f] * W[f * Dd + d];
  h[idx] = s;
}

// ---------------------------------------------------------------------------
// Residual + LayerNorm: out = LN(x + a (+ b)) * g + beta.  One row per block.
// The x/a rows are staged LDS-side with gfx1250 async global->LDS copies
// (GLOBAL_LOAD_ASYNC_TO_LDS_B32, tracked by ASYNCcnt).
// ---------------------------------------------------------------------------
__global__ __launch_bounds__(128) void k_add_ln(
    const float* __restrict__ x, const float* __restrict__ a,
    const float* __restrict__ b, const float* __restrict__ g,
    const float* __restrict__ beta, float* __restrict__ out) {
  const int row = blockIdx.x;
  const int t = threadIdx.x;            // 128 threads == Dd
  __shared__ float lx[Dd], la[Dd];
  __shared__ float red[Dd];
  const size_t idx = (size_t)row * Dd + t;

  // Async DMA: each lane copies one dword from global memory into LDS (ISA 15.18.3 op 96).
  {
    const float* gx = x + idx;
    const float* ga = a + idx;
    const unsigned dx = lds_off32(&lx[t]);
    const unsigned da = lds_off32(&la[t]);
    asm volatile("global_load_async_to_lds_b32 %0, %1, off"
                 :: "v"(dx), "v"(gx) : "memory");
    asm volatile("global_load_async_to_lds_b32 %0, %1, off"
                 :: "v"(da), "v"(ga) : "memory");
    asm volatile("s_wait_asynccnt 0x0" ::: "memory");
  }
  __syncthreads();

  float val = lx[t] + la[t] + (b ? b[idx] : 0.0f);
  red[t] = val;
  __syncthreads();
  for (int off = 64; off > 0; off >>= 1) {
    if (t < off) red[t] += red[t + off];
    __syncthreads();
  }
  const float mu = red[0] * (1.0f / Dd);
  __syncthreads();
  const float d = val - mu;
  red[t] = d * d;
  __syncthreads();
  for (int off = 64; off > 0; off >>= 1) {
    if (t < off) red[t] += red[t + off];
    __syncthreads();
  }
  const float var = red[0] * (1.0f / Dd);
  out[idx] = d * rsqrtf(var + 1e-5f) * g[t] + beta[t];
}

// ---------------------------------------------------------------------------
// z[(b*N+n), t, :] = h[(b*T+t)*N+n, :] + temporal_pos[t, :]
// ---------------------------------------------------------------------------
__global__ void k_build_z(const float* __restrict__ h, const float* __restrict__ tpos,
                          float* __restrict__ z) {
  const size_t idx = (size_t)blockIdx.x * blockDim.x + threadIdx.x;
  if (idx >= (size_t)NT * Dd) return;
  const int d = (int)(idx & (Dd - 1));
  const size_t row = idx >> 7;              // (b*T+t)*N + n
  const int n = (int)(row % Nn);
  const size_t btq = row / Nn;
  const int t = (int)(btq % Tn);
  const int bb = (int)(btq / Tn);
  const size_t zrow = (size_t)(bb * Nn + n) * Tn + t;
  z[zrow * Dd + d] = h[idx] + tpos[t * Dd + d];
}

// ---------------------------------------------------------------------------
// Spatial attention (dense 64x64 per (b*t, head), diagonal masked).
// alpha_ij = (q_i.k_j + ea_ij*(q_i.We_h) + q_i.be_h)*scale
// agg_i    = sum_j w_ij v_j + (sum_j w_ij ea_ij)*We_h + be_h
// The EA tile is pre-staged with coalesced loads (row j of ea is 63
// contiguous floats), so the WMMA epilogue is branchless.
// ---------------------------------------------------------------------------
__global__ __launch_bounds__(128) void k_spatial_attn(
    const float* __restrict__ q, const float* __restrict__ k,
    const float* __restrict__ v, const float* __restrict__ ea,
    const float* __restrict__ We, const float* __restrict__ be,
    float* __restrict__ agg) {
  const int bt = blockIdx.x >> 2;
  const int hh = blockIdx.x & 3;
  const int tid = threadIdx.x, lane = tid & 31, wave = tid >> 5;
  const int base = bt * Nn;
  const int lk0 = (lane & 16) ? 8 : 0;

  __shared__ _Float16 lq[Nn][Cc], lk[Nn][Cc];
  __shared__ _Float16 lvT[Cc][Nn];          // transposed V for contiguous B-frags
  __shared__ float Sm[Nn][Nn];
  __shared__ float EAm[Nn][Nn];
  __shared__ _Float16 P[Nn][Nn];
  __shared__ float qWe[Nn], qBe[Nn], rsum[Nn];

  for (int idx = tid; idx < Nn * Cc; idx += 128) {
    const int node = idx >> 5, c = idx & 31;
    const size_t g = (size_t)(base + node) * Dd + hh * Cc + c;
    lq[node][c] = (_Float16)q[g];
    lk[node][c] = (_Float16)k[g];
    lvT[c][node] = (_Float16)v[g];
  }
  // EA tile: coalesced over the edge list; EAm[i][j] = ea(src=j -> dst=i).
  for (int e = tid; e < Nn * (Nn - 1); e += 128) {
    const int j = e / (Nn - 1);
    const int r = e - j * (Nn - 1);
    const int i = r + (r >= j ? 1 : 0);
    EAm[i][j] = ea[(size_t)bt * En + e];
  }
  if (tid < Nn) {
    EAm[tid][tid] = 0.0f;
    float s1 = 0.0f, s2 = 0.0f;
    const size_t g = (size_t)(base + tid) * Dd + hh * Cc;
    for (int c = 0; c < Cc; ++c) {
      const float qv = q[g + c];
      s1 += qv * We[hh * Cc + c];
      s2 += qv * be[hh * Cc + c];
    }
    qWe[tid] = s1; qBe[tid] = s2;
  }
  __syncthreads();

  // S = Q K^T (WMMA), with rank-1 edge bias and diagonal mask (branchless).
  {
    const int tm = wave * 16;
    const int mrow = tm + (lane & 15);
    const v16h a = frag16(&lq[mrow][0], lk0);
    for (int jt = 0; jt < 4; ++jt) {
      const int ncol = jt * 16 + (lane & 15);
      const v16h b = frag16(&lk[ncol][0], lk0);
      v8f c8 = {};
      c8 = __builtin_amdgcn_wmma_f32_16x16x32_f16(false, a, false, b,
                                                  (short)0, c8, false, false);
#pragma unroll
      for (int r = 0; r < 8; ++r) {
        const int i = tm + r + ((lane & 16) ? 8 : 0);
        const float eav = EAm[i][ncol];
        const float al = (c8[r] + eav * qWe[i] + qBe[i]) * SCALE;
        Sm[i][ncol] = (i == ncol) ? -1e30f : al;
      }
    }
  }
  __syncthreads();

  // Row softmax + weighted-edge sum (threads 0..63 each own one row).
  if (tid < Nn) {
    const int i = tid;
    float m = -1e30f;
    for (int j = 0; j < Nn; ++j) m = fmaxf(m, Sm[i][j]);
    float ss = 0.0f;
    for (int j = 0; j < Nn; ++j) { const float e = expf(Sm[i][j] - m); Sm[i][j] = e; ss += e; }
    const float inv = 1.0f / ss;
    float rs = 0.0f;
    for (int j = 0; j < Nn; ++j) {
      const float w = Sm[i][j] * inv;
      P[i][j] = (_Float16)w;
      rs += w * EAm[i][j];
    }
    rsum[i] = rs;
  }
  __syncthreads();

  // O = P @ V (WMMA, K=64 -> two K-steps), plus rank-1 correction.
  {
    const int tm = wave * 16;
    const int mrow = tm + (lane & 15);
    for (int ct = 0; ct < 2; ++ct) {
      const int ncol = ct * 16 + (lane & 15);
      v8f c8 = {};
      for (int ks = 0; ks < 2; ++ks) {
        const v16h a = frag16(&P[mrow][ks * 32], lk0);
        const v16h b = frag16(&lvT[ncol][ks * 32], lk0);
        c8 = __builtin_amdgcn_wmma_f32_16x16x32_f16(false, a, false, b,
                                                    (short)0, c8, false, false);
      }
#pragma unroll
      for (int r = 0; r < 8; ++r) {
        const int i = tm + r + ((lane & 16) ? 8 : 0);
        const float val = c8[r] + rsum[i] * We[hh * Cc + ncol] + be[hh * Cc + ncol];
        agg[(size_t)(base + i) * Dd + hh * Cc + ncol] = val;
      }
    }
  }
}

// ---------------------------------------------------------------------------
// Temporal causal attention: one block per sequence, one wave per head.
// ---------------------------------------------------------------------------
__global__ __launch_bounds__(128) void k_temporal_attn(
    const float* __restrict__ qkv, float* __restrict__ attn) {
  const int s = blockIdx.x;                       // 0..511
  const int tid = threadIdx.x, lane = tid & 31, hh = tid >> 5;
  const int lk0 = (lane & 16) ? 8 : 0;

  __shared__ _Float16 lq[Hh][Tn][Cc], lk[Hh][Tn][Cc];
  __shared__ _Float16 lvT[Hh][Cc][Tn];
  __shared__ float Sm[Hh][Tn][Tn];
  __shared__ _Float16 P[Hh][Tn][Tn];

  {
    const int t = lane;
    const size_t rowb = ((size_t)s * Tn + t) * (3 * Dd);
    for (int c = 0; c < Cc; ++c) {
      lq[hh][t][c] = (_Float16)qkv[rowb + hh * Cc + c];
      lk[hh][t][c] = (_Float16)qkv[rowb + Dd + hh * Cc + c];
      lvT[hh][c][t] = (_Float16)qkv[rowb + 2 * Dd + hh * Cc + c];
    }
  }
  __syncthreads();

  for (int it = 0; it < 2; ++it) {
    const int mrow = it * 16 + (lane & 15);
    const v16h a = frag16(&lq[hh][mrow][0], lk0);
    for (int jt = 0; jt < 2; ++jt) {
      const int ncol = jt * 16 + (lane & 15);
      const v16h b = frag16(&lk[hh][ncol][0], lk0);
      v8f c8 = {};
      c8 = __builtin_amdgcn_wmma_f32_16x16x32_f16(false, a, false, b,
                                                  (short)0, c8, false, false);
#pragma unroll
      for (int r = 0; r < 8; ++r) {
        const int i = it * 16 + r + ((lane & 16) ? 8 : 0);
        Sm[hh][i][ncol] = (ncol <= i) ? c8[r] * SCALE : -1e30f;
      }
    }
  }
  __syncthreads();

  {
    const int i = lane;                           // 32 lanes <-> 32 rows
    float m = -1e30f;
    for (int j = 0; j < Tn; ++j) m = fmaxf(m, Sm[hh][i][j]);
    float ss = 0.0f;
    for (int j = 0; j < Tn; ++j) { const float e = expf(Sm[hh][i][j] - m); Sm[hh][i][j] = e; ss += e; }
    const float inv = 1.0f / ss;
    for (int j = 0; j < Tn; ++j) P[hh][i][j] = (_Float16)(Sm[hh][i][j] * inv);
  }
  __syncthreads();

  for (int it = 0; it < 2; ++it) {
    const int mrow = it * 16 + (lane & 15);
    const v16h a = frag16(&P[hh][mrow][0], lk0);
    for (int ct = 0; ct < 2; ++ct) {
      const int ncol = ct * 16 + (lane & 15);
      const v16h b = frag16(&lvT[hh][ncol][0], lk0);
      v8f c8 = {};
      c8 = __builtin_amdgcn_wmma_f32_16x16x32_f16(false, a, false, b,
                                                  (short)0, c8, false, false);
#pragma unroll
      for (int r = 0; r < 8; ++r) {
        const int i = it * 16 + r + ((lane & 16) ? 8 : 0);
        attn[((size_t)s * Tn + i) * Dd + hh * Cc + ncol] = c8[r];
      }
    }
  }
}

// ---------------------------------------------------------------------------
extern "C" void kernel_launch(void* const* d_in, const int* in_sizes, int n_in,
                              void* d_out, int out_size, void* d_ws, size_t ws_size,
                              hipStream_t stream) {
  (void)in_sizes; (void)n_in; (void)out_size; (void)ws_size;
  const float* input_frames = (const float*)d_in[0];
  const float* edge_attr    = (const float*)d_in[1];
  const float* proj_W  = (const float*)d_in[2];
  const float* proj_b  = (const float*)d_in[3];
  const float* sp_Wq   = (const float*)d_in[4];
  const float* sp_bq   = (const float*)d_in[5];
  const float* sp_Wk   = (const float*)d_in[6];
  const float* sp_bk   = (const float*)d_in[7];
  const float* sp_Wv   = (const float*)d_in[8];
  const float* sp_bv   = (const float*)d_in[9];
  const float* sp_We   = (const float*)d_in[10];
  const float* sp_be   = (const float*)d_in[11];
  const float* sp_Ws   = (const float*)d_in[12];
  const float* sp_bs   = (const float*)d_in[13];
  const float* sp_lng  = (const float*)d_in[14];
  const float* sp_lnb  = (const float*)d_in[15];
  const float* tpos    = (const float*)d_in[16];
  const float* t_Wqkv  = (const float*)d_in[17];
  const float* t_bqkv  = (const float*)d_in[18];
  const float* t_Wo    = (const float*)d_in[19];
  const float* t_bo    = (const float*)d_in[20];
  const float* t_ln1g  = (const float*)d_in[21];
  const float* t_ln1b  = (const float*)d_in[22];
  const float* t_W1    = (const float*)d_in[23];
  const float* t_b1    = (const float*)d_in[24];
  const float* t_W2    = (const float*)d_in[25];
  const float* t_b2    = (const float*)d_in[26];
  const float* t_ln2g  = (const float*)d_in[27];
  const float* t_ln2b  = (const float*)d_in[28];
  const float* head_W1 = (const float*)d_in[29];
  const float* head_b1 = (const float*)d_in[30];
  const float* head_W2 = (const float*)d_in[31];
  const float* head_b2 = (const float*)d_in[32];
  // d_in[33] (edge_index) not needed: complete-graph structure is hardcoded.

  float* wsf = (float*)d_ws;
  const size_t SZ = (size_t)NT * Dd;          // 2,097,152 floats
  float* h    = wsf + 0 * SZ;
  float* qb   = wsf + 1 * SZ;
  float* kb   = wsf + 2 * SZ;
  float* vb   = wsf + 3 * SZ;
  float* skip = wsf + 4 * SZ;
  float* agg  = wsf + 5 * SZ;
  float* z    = wsf + 6 * SZ;
  float* qkv  = wsf + 7 * SZ;                 // 3*SZ (N=384)
  // Reuse after z is built (h/qb/kb/vb/skip/agg become free):
  float* attn = wsf + 0 * SZ;
  float* oprj = wsf + 1 * SZ;
  float* ffn  = wsf + 2 * SZ;                 // 4*SZ (N=512)
  float* hid  = wsf + 2 * SZ;                 // head stage

  auto gemm = [&](const float* A, int lda, const float* W, const float* bias,
                  float* out, int ldo, int M, int N, int K, int act) {
    const int blocks = (M / 64) * ((N + 63) / 64);
    k_gemm<false><<<blocks, 256, 0, stream>>>(A, lda, W, bias, out, ldo, M, N, K, act);
  };

  k_proj<<<(NT * Dd) / 256, 256, 0, stream>>>(input_frames, proj_W, proj_b, h);

  for (int l = 0; l < 3; ++l) {
    const size_t wo = (size_t)l * Dd * Dd;
    gemm(h, Dd, sp_Wq + wo, sp_bq + l * Dd, qb,   Dd, NT, Dd, Dd, 0);
    gemm(h, Dd, sp_Wk + wo, sp_bk + l * Dd, kb,   Dd, NT, Dd, Dd, 0);
    gemm(h, Dd, sp_Wv + wo, sp_bv + l * Dd, vb,   Dd, NT, Dd, Dd, 0);
    gemm(h, Dd, sp_Ws + wo, sp_bs + l * Dd, skip, Dd, NT, Dd, Dd, 0);
    k_spatial_attn<<<Bn * Tn * Hh, 128, 0, stream>>>(
        qb, kb, vb, edge_attr, sp_We + (size_t)l * Dd, sp_be + (size_t)l * Dd, agg);
    k_add_ln<<<NT, Dd, 0, stream>>>(h, agg, skip, sp_lng + l * Dd, sp_lnb + l * Dd, h);
  }

  k_build_z<<<(NT * Dd) / 256, 256, 0, stream>>>(h, tpos, z);

  for (int l = 0; l < 3; ++l) {
    gemm(z, Dd, t_Wqkv + (size_t)l * Dd * 3 * Dd, t_bqkv + l * 3 * Dd,
         qkv, 3 * Dd, NT, 3 * Dd, Dd, 0);
    k_temporal_attn<<<SEQ, 128, 0, stream>>>(qkv, attn);
    gemm(attn, Dd, t_Wo + (size_t)l * Dd * Dd, t_bo + l * Dd, oprj, Dd, NT, Dd, Dd, 0);
    k_add_ln<<<NT, Dd, 0, stream>>>(z, oprj, nullptr, t_ln1g + l * Dd, t_ln1b + l * Dd, z);
    gemm(z, Dd, t_W1 + (size_t)l * Dd * 4 * Dd, t_b1 + l * 4 * Dd,
         ffn, 4 * Dd, NT, 4 * Dd, Dd, 1);
    gemm(ffn, 4 * Dd, t_W2 + (size_t)l * 4 * Dd * Dd, t_b2 + l * Dd,
         oprj, Dd, NT, Dd, 4 * Dd, 0);
    k_add_ln<<<NT, Dd, 0, stream>>>(z, oprj, nullptr, t_ln2g + l * Dd, t_ln2b + l * Dd, z);
  }

  // Head: rows are z[s, T-1, :] (strided A), then GELU MLP to F=6.
  gemm(z + (size_t)(Tn - 1) * Dd, Tn * Dd, head_W1, head_b1, hid, Dd, SEQ, Dd, Dd, 1);
  k_gemm<true><<<(SEQ / 64) * 1, 256, 0, stream>>>(
      hid, Dd, head_W2, head_b2, (float*)d_out, Fd, SEQ, Fd, Dd, 0);
}
